// LSTM_618475291240
// MI455X (gfx1250) — matrix operand order repaired and checked by reading, not compiled, not run on hardware
//
#include <hip/hip_runtime.h>

typedef __attribute__((ext_vector_type(16))) __bf16 v16bf;
typedef __attribute__((ext_vector_type(8)))  float  v8f;

#define BATCH  64
#define SEQ    512
#define INPUT  512
#define HIDDEN 1024
#define KTOT   (INPUT + HIDDEN)   // 1536
#define FOURH  (4 * HIDDEN)       // 4096
#define LDS_STRIDE (KTOT + 16)    // 1552 bf16 per row: stagger LDS banks
#define NBLOCKS 32                // 4 mtiles x 8 hgroups; 8 waves/block

union FragPack { uint4 u[2]; v16bf v; };

// ---------------- prep: f32 -> bf16 weight copy (gate-concatenated) + bias ----
__global__ void prep_weights(const float* __restrict__ Wf, const float* __restrict__ Wi,
                             const float* __restrict__ Wo, const float* __restrict__ Wg,
                             const float* __restrict__ bf_, const float* __restrict__ bi_,
                             const float* __restrict__ bo_, const float* __restrict__ bg_,
                             __bf16* __restrict__ Wbf, float* __restrict__ bias) {
  long idx = (long)blockIdx.x * blockDim.x + threadIdx.x;
  if (idx < (long)FOURH * KTOT) {
    int row  = (int)(idx / KTOT);
    int col  = (int)(idx - (long)row * KTOT);
    int gate = row >> 10, hrow = row & (HIDDEN - 1);
    const float* src = (gate == 0) ? Wf : (gate == 1) ? Wi : (gate == 2) ? Wo : Wg;
    Wbf[idx] = (__bf16)src[(long)hrow * KTOT + col];
  }
  if (idx < FOURH) {
    int gate = (int)(idx >> 10), h = (int)(idx & (HIDDEN - 1));
    const float* b = (gate == 0) ? bf_ : (gate == 1) ? bi_ : (gate == 2) ? bo_ : bg_;
    bias[idx] = b[h];
  }
}

// ---------------- prep: embedding gather -> bf16 xs [B,S,I] ------------------
__global__ void prep_embed(const int* __restrict__ X, const float* __restrict__ emb,
                           __bf16* __restrict__ Xemb) {
  long idx = (long)blockIdx.x * blockDim.x + threadIdx.x;
  if (idx < (long)BATCH * SEQ * INPUT) {
    int  i  = (int)(idx & (INPUT - 1));
    long bs = idx >> 9;               // / INPUT
    int  tok = X[bs];
    Xemb[idx] = (__bf16)emb[(long)tok * INPUT + i];
  }
}

// ---------------- prep: h0 (bf16) / c0 (f32), zero the sync counter ----------
__global__ void init_state(const float* __restrict__ h0, const float* __restrict__ c0,
                           __bf16* __restrict__ hbuf, float* __restrict__ cbuf,
                           unsigned* __restrict__ sync) {
  int idx = blockIdx.x * blockDim.x + threadIdx.x;
  if (idx == 0) *sync = 0u;
  if (idx < BATCH * HIDDEN) {
    hbuf[idx] = (__bf16)h0[idx];
    cbuf[idx] = c0[idx];
  }
}

// ---------------- persistent LSTM scan ---------------------------------------
// One launch; t-loop on device with grid-wide split-phase sync (monotonic L2
// atomic counter). 32 blocks x 256 threads: block -> (mtile = bx>>3, hgroup =
// bx&7); wave w owns htile = hgroup*8 + w, a 16(batch) x 16(H) output slice,
// computing all four gate tiles so the cell update stays in registers.
__global__ __launch_bounds__(256, 1)
void lstm_persist(const __bf16* __restrict__ Xemb,
                  __bf16* __restrict__ h0buf, __bf16* __restrict__ h1buf,
                  float* __restrict__ cstate,
                  const __bf16* __restrict__ Wbf, const float* __restrict__ bias,
                  float* __restrict__ latent, unsigned* __restrict__ sync) {
  __shared__ __bf16 sA[16 * LDS_STRIDE];

  const int tid   = threadIdx.x;
  const int lane  = tid & 31;
  const int wave  = tid >> 5;
  const int mtile = blockIdx.x >> 3;
  const int htile = ((blockIdx.x & 7) << 3) | wave;
  const int b0    = mtile * 16;

  // ---- t-invariant setup (hoisted out of the scan) ----
  // A fragment (16-bit 16x32): row = lane&15; lanes<16 take K
  // 0..15, lanes>=16 take 16..31 (matches B).
  const int     arow  = lane & 15;
  const int     ahalf = (lane >> 4) << 4;           // 0 or 16 elements
  const __bf16* aBase = &sA[arow * LDS_STRIDE + ahalf];

  // B fragment: lane holds column n = htile*16 + (lane&15); lanes<16 carry
  // K=kb..kb+15, lanes>=16 carry K=kb+16..kb+31 (W rows are K-contiguous).
  const int     ncol = htile * 16 + (lane & 15);
  const int     kb0  = (lane >> 4) << 4;            // 0 or 16 elements
  const __bf16* bF = Wbf + (long)(0 * HIDDEN + ncol) * KTOT + kb0;
  const __bf16* bI = Wbf + (long)(1 * HIDDEN + ncol) * KTOT + kb0;
  const __bf16* bO = Wbf + (long)(2 * HIDDEN + ncol) * KTOT + kb0;
  const __bf16* bG = Wbf + (long)(3 * HIDDEN + ncol) * KTOT + kb0;

  const int   h  = htile * 16 + (lane & 15);
  const float Bf = bias[h];
  const float Bi = bias[HIDDEN + h];
  const float Bo = bias[2 * HIDDEN + h];
  const float Bg = bias[3 * HIDDEN + h];

  for (int t = 0; t < SEQ; ++t) {
    const __bf16* hprev = (t & 1) ? h1buf : h0buf;
    __bf16*       hnext = (t & 1) ? h0buf : h1buf;

    // ---- stage combined[b0..b0+15][0..1535] into LDS (bf16, 16B chunks) ----
    // 192 chunks/row: 64 from x_t (512 bf16), 128 from h_{t-1} (1024 bf16)
    for (int c = tid; c < 16 * 192; c += 256) {
      int row = c / 192, ch = c - row * 192;
      uint4 v;
      if (ch < 64)
        v = *((const uint4*)(Xemb + ((long)(b0 + row) * SEQ + t) * INPUT) + ch);
      else
        v = *((const uint4*)(hprev + (long)(b0 + row) * HIDDEN) + (ch - 64));
      *(uint4*)(&sA[row * LDS_STRIDE + ch * 8]) = v;
    }
    __syncthreads();

    v8f accf = {}, acci = {}, acco = {}, accg = {};

    for (int kc = 0; kc < 48; ++kc) {
      const int kb = kc * 32;
      FragPack a, f_, i_, o_, g_;
      a.u[0]  = *(const uint4*)(aBase + kb);
      a.u[1]  = *(const uint4*)(aBase + kb + 8);
      f_.u[0] = *(const uint4*)(bF + kb);  f_.u[1] = *(const uint4*)(bF + kb + 8);
      i_.u[0] = *(const uint4*)(bI + kb);  i_.u[1] = *(const uint4*)(bI + kb + 8);
      o_.u[0] = *(const uint4*)(bO + kb);  o_.u[1] = *(const uint4*)(bO + kb + 8);
      g_.u[0] = *(const uint4*)(bG + kb);  g_.u[1] = *(const uint4*)(bG + kb + 8);

      if (kc + 2 < 48) {                    // warm L0 for the next-next K chunk
        __builtin_prefetch(bF + kb + 64, 0, 1);
        __builtin_prefetch(bG + kb + 64, 0, 1);
      }

      accf = __builtin_amdgcn_wmma_f32_16x16x32_bf16(false, a.v, false, f_.v, (short)0, accf, false, false);
      acci = __builtin_amdgcn_wmma_f32_16x16x32_bf16(false, a.v, false, i_.v, (short)0, acci, false, false);
      acco = __builtin_amdgcn_wmma_f32_16x16x32_bf16(false, a.v, false, o_.v, (short)0, acco, false, false);
      accg = __builtin_amdgcn_wmma_f32_16x16x32_bf16(false, a.v, false, g_.v, (short)0, accg, false, false);
    }

    // ---- gate math + cell update, fully in registers ----
#pragma unroll
    for (int r = 0; r < 8; ++r) {
      const int  mm = r + ((lane >> 4) << 3);   // C/D layout: M = r (+8 hi lanes)
      const int  b  = b0 + mm;
      const long ci = (long)b * HIDDEN + h;

      float zf = accf[r] + Bf;
      float zi = acci[r] + Bi;
      float zo = acco[r] + Bo;
      float zg = accg[r] + Bg;

      float ft = 1.f / (1.f + __expf(-zf));
      float it = 1.f / (1.f + __expf(-zi));
      float ot = 1.f / (1.f + __expf(-zo));
      float eg = __expf(2.f * zg);
      float gt = (eg - 1.f) / (eg + 1.f);

      float c_t = ft * cstate[ci] + it * gt;
      float ec  = __expf(2.f * c_t);
      float h_t = ot * ((ec - 1.f) / (ec + 1.f));

      cstate[ci] = c_t;
      latent[((long)b * SEQ + t) * HIDDEN + h] = h_t;
      hnext[ci] = (__bf16)h_t;
    }

    // ---- grid-wide split-phase sync (monotonic counter, no reset races) ----
    if (t + 1 < SEQ) {
      __syncthreads();                           // block done reading sA / accs
      if (tid == 0) {
        __builtin_amdgcn_fence(__ATOMIC_RELEASE, "agent");   // flush h/c writes
        __hip_atomic_fetch_add(sync, 1u, __ATOMIC_RELAXED, __HIP_MEMORY_SCOPE_AGENT);
        const unsigned target = (unsigned)NBLOCKS * (unsigned)(t + 1);
        while (__hip_atomic_load(sync, __ATOMIC_RELAXED, __HIP_MEMORY_SCOPE_AGENT) < target) {
          __builtin_amdgcn_s_sleep(2);
        }
      }
      __syncthreads();
      __builtin_amdgcn_fence(__ATOMIC_ACQUIRE, "agent");     // invalidate stale h
    }
  }
}

// ---------------- finalize: append (h_f, c_f) after latent -------------------
__global__ void finalize(float* __restrict__ out, const float* __restrict__ cstate) {
  int idx = blockIdx.x * blockDim.x + threadIdx.x;
  if (idx < BATCH * HIDDEN) {
    int  b = idx >> 10, h = idx & (HIDDEN - 1);
    long base = (long)BATCH * SEQ * HIDDEN;
    out[base + idx] = out[((long)b * SEQ + (SEQ - 1)) * HIDDEN + h];  // h_f
    out[base + BATCH * HIDDEN + idx] = cstate[idx];                   // c_f
  }
}

// ---------------- host orchestration -----------------------------------------
extern "C" void kernel_launch(void* const* d_in, const int* in_sizes, int n_in,
                              void* d_out, int out_size, void* d_ws, size_t ws_size,
                              hipStream_t stream) {
  (void)in_sizes; (void)n_in; (void)out_size; (void)ws_size;
  const int*   X    = (const int*)d_in[0];
  const float* h0   = (const float*)d_in[1];
  const float* c0   = (const float*)d_in[2];
  const float* emb  = (const float*)d_in[3];
  const float* Wf_w = (const float*)d_in[4];
  const float* Wf_b = (const float*)d_in[5];
  const float* Wi_w = (const float*)d_in[6];
  const float* Wi_b = (const float*)d_in[7];
  const float* Wo_w = (const float*)d_in[8];
  const float* Wo_b = (const float*)d_in[9];
  const float* Wg_w = (const float*)d_in[10];
  const float* Wg_b = (const float*)d_in[11];
  float* out = (float*)d_out;
  char*  ws  = (char*)d_ws;

  // workspace layout (bytes)
  __bf16*   Wbf  = (__bf16*)  (ws);                 // 4096*1536*2 = 12,582,912
  float*    bias = (float*)   (ws + 12582912);      // 16,384
  __bf16*   Xemb = (__bf16*)  (ws + 12599296);      // 64*512*512*2 = 33,554,432
  __bf16*   hA   = (__bf16*)  (ws + 46153728);      // 131,072
  __bf16*   hB   = (__bf16*)  (ws + 46284800);      // 131,072
  float*    cst  = (float*)   (ws + 46415872);      // 262,144
  unsigned* sync = (unsigned*)(ws + 46678016);      // 4 (+pad)

  prep_weights<<<((long)FOURH * KTOT + 255) / 256, 256, 0, stream>>>(
      Wf_w, Wi_w, Wo_w, Wg_w, Wf_b, Wi_b, Wo_b, Wg_b, Wbf, bias);
  prep_embed<<<((long)BATCH * SEQ * INPUT + 255) / 256, 256, 0, stream>>>(X, emb, Xemb);
  init_state<<<(BATCH * HIDDEN + 255) / 256, 256, 0, stream>>>(h0, c0, hA, cst, sync);

  lstm_persist<<<NBLOCKS, 256, 0, stream>>>(Xemb, hA, hB, cst, Wbf, bias, out, sync);

  finalize<<<(BATCH * HIDDEN + 255) / 256, 256, 0, stream>>>(out, cst);
}
